// Attention_48825188221088
// MI455X (gfx1250) — compile-verified, matrix operand order checked by reading, not run on hardware
//
#include <hip/hip_runtime.h>
#include <hip/hip_bf16.h>

typedef __attribute__((ext_vector_type(16))) _Float16 v16h;
typedef __attribute__((ext_vector_type(8)))  _Float16 v8h;
typedef __attribute__((ext_vector_type(4)))  _Float16 v4h;
typedef __attribute__((ext_vector_type(8)))  float    v8f;
typedef __attribute__((ext_vector_type(4)))  float    v4f;

#define NB 8
#define NS 2048
#define NH 1024

union ABPack { v16h v; v8h h[2]; };

// convert 8 consecutive f32 (16B-aligned) to v8h
__device__ __forceinline__ v8h cvt8(const float* __restrict__ p) {
    v4f a = *(const v4f*)p;
    v4f b = *(const v4f*)(p + 4);
    v8h r;
    r[0] = (_Float16)a[0]; r[1] = (_Float16)a[1];
    r[2] = (_Float16)a[2]; r[3] = (_Float16)a[3];
    r[4] = (_Float16)b[0]; r[5] = (_Float16)b[1];
    r[6] = (_Float16)b[2]; r[7] = (_Float16)b[3];
    return r;
}

__device__ __forceinline__ v8f wmma16(const ABPack& a, const ABPack& b, v8f c) {
    return __builtin_amdgcn_wmma_f32_16x16x32_f16(false, a.v, false, b.v,
                                                  (short)0, c, false, false);
}

__device__ __forceinline__ void ldpack(ABPack& p, const _Float16* base, int kk, int koff) {
    p.h[0] = *(const v8h*)(base + kk + koff);
    p.h[1] = *(const v8h*)(base + kk + 16 + koff);
}

// ---------------------------------------------------------------------------
// Kernel 0: bulk f32 -> f16 conversion (vectorized 4-wide)
// ---------------------------------------------------------------------------
__global__ __launch_bounds__(256) void cvt_kernel(
    const float* __restrict__ src, _Float16* __restrict__ dst, int n4)
{
    const int i = blockIdx.x * 256 + threadIdx.x;
    if (i < n4) {
        v4f a = ((const v4f*)src)[i];
        v4h r;
        r[0] = (_Float16)a[0]; r[1] = (_Float16)a[1];
        r[2] = (_Float16)a[2]; r[3] = (_Float16)a[3];
        ((v4h*)dst)[i] = r;
    }
}

// ---------------------------------------------------------------------------
// Kernel 1: out = X @ W^T + b   (f16 operands, f16 out)   [B*S, H]
// One wave per 32x64 output block (2x4 register tile of 16x16 wmma tiles):
// each k-step loads 2 A packs + 4 B packs and issues 8 independent wmmas.
// ---------------------------------------------------------------------------
__global__ __launch_bounds__(256) void linear_qk_kernel(
    const _Float16* __restrict__ xh,   // [B*S, H] f16
    const _Float16* __restrict__ wh,   // [H, H] f16
    const float*    __restrict__ bias, // [H] f32
    _Float16* __restrict__ qk)         // [B*S, H] f16
{
    const int lane = threadIdx.x & 31;
    const int wave = threadIdx.x >> 5;
    const int tile = blockIdx.x * 8 + wave;     // (B*S/32)*(H/64) = 8192 tiles
    const int mt   = tile >> 4;                 // 0..511
    const int nt   = tile & 15;                 // 0..15
    const int m    = lane & 15;
    const int hi   = lane >> 4;
    const int koff = hi * 8;

    const _Float16* arow0 = xh + (size_t)(mt * 32 + m) * NH;
    const _Float16* arow1 = xh + (size_t)(mt * 32 + 16 + m) * NH;
    const _Float16* brow[4];
#pragma unroll
    for (int j = 0; j < 4; ++j)
        brow[j] = wh + (size_t)(nt * 64 + j * 16 + m) * NH;

    v8f c0[4] = {}, c1[4] = {};
    for (int kk = 0; kk < NH; kk += 32) {
        ABPack a0, a1, bb[4];
        ldpack(a0, arow0, kk, koff);
        ldpack(a1, arow1, kk, koff);
#pragma unroll
        for (int j = 0; j < 4; ++j) ldpack(bb[j], brow[j], kk, koff);
#pragma unroll
        for (int j = 0; j < 4; ++j) {
            c0[j] = wmma16(a0, bb[j], c0[j]);
            c1[j] = wmma16(a1, bb[j], c1[j]);
        }
    }
    const int n = lane & 15;
    float bv[4];
#pragma unroll
    for (int j = 0; j < 4; ++j) bv[j] = bias[nt * 64 + j * 16 + n];
#pragma unroll
    for (int r = 0; r < 8; ++r) {
        const size_t row0 = (size_t)(mt * 32 + r + hi * 8) * NH;
        const size_t row1 = (size_t)(mt * 32 + 16 + r + hi * 8) * NH;
#pragma unroll
        for (int j = 0; j < 4; ++j) {
            qk[row0 + nt * 64 + j * 16 + n] = (_Float16)(c0[j][r] + bv[j]);
            qk[row1 + nt * 64 + j * 16 + n] = (_Float16)(c1[j][r] + bv[j]);
        }
    }
}

// ---------------------------------------------------------------------------
// Kernel 2: Vt[b][h][s] = (f16) X[b][s][h]   (LDS tiled transpose)
// ---------------------------------------------------------------------------
__global__ __launch_bounds__(256) void vt_kernel(
    const float* __restrict__ x, _Float16* __restrict__ vt)
{
    __shared__ float tile[32][33];
    const int b  = blockIdx.z;
    const int s0 = blockIdx.x * 32;
    const int h0 = blockIdx.y * 32;
    const int tx = threadIdx.x, ty = threadIdx.y;
    const float* src = x + (size_t)b * NS * NH;
    _Float16*    dst = vt + (size_t)b * NH * NS;
#pragma unroll
    for (int j = 0; j < 32; j += 8)
        tile[ty + j][tx] = src[(size_t)(s0 + ty + j) * NH + h0 + tx];
    __syncthreads();
#pragma unroll
    for (int j = 0; j < 32; j += 8)
        dst[(size_t)(h0 + ty + j) * NS + s0 + tx] = (_Float16)tile[tx][ty + j];
}

// ---------------------------------------------------------------------------
// Kernel 3: per (batch, 16-query tile): S = Q K^T ; P = softmax(S) ; O = P V
// 8 waves / 256 threads per block. 160KB dynamic LDS -> 2 blocks per WGP.
// ---------------------------------------------------------------------------
__global__ __launch_bounds__(256) void attn_kernel(
    const _Float16* __restrict__ qk,  // [B, S, H] f16 (Q == K)
    const _Float16* __restrict__ vt,  // [B, H, S] f16 (V transposed)
    float* __restrict__ out)          // [B, S, H] f32
{
    extern __shared__ __align__(16) char smem[];
    _Float16* qs = (_Float16*)smem;                 // 16*1024 f16 = 32KB
    float*    sc = (float*)(smem + 16 * NH * 2);    // 16*2048 f32 = 128KB (scores, then P)

    const int lane = threadIdx.x & 31;
    const int wave = threadIdx.x >> 5;
    const int b    = blockIdx.x >> 7;    // /128
    const int qt   = blockIdx.x & 127;
    const int q0   = qt * 16;
    const int m    = lane & 15;
    const int hi   = lane >> 4;
    const int koff = hi * 8;

    // ---- Phase 0: stage Q tile (16 rows x 1024) into LDS ----
    {
        const uint4* src  = (const uint4*)(qk + ((size_t)b * NS + q0) * NH);
        uint4*       dst4 = (uint4*)qs;
        for (int i = threadIdx.x; i < 16 * NH / 8; i += 256) dst4[i] = src[i];
    }
    __syncthreads();

    // ---- Phase 1: scores S[16][2048] = Q K^T.
    // Each wave: 16 key tiles, 4 at a time (A from LDS loaded once per k-step).
    const _Float16* qrow = qs + (size_t)m * NH;
    for (int g = 0; g < 4; ++g) {
        const int kt0 = wave * 16 + g * 4;
        const _Float16* krow[4];
#pragma unroll
        for (int j = 0; j < 4; ++j)
            krow[j] = qk + ((size_t)b * NS + (kt0 + j) * 16 + m) * NH;
        v8f c[4] = {};
        for (int kk = 0; kk < NH; kk += 32) {
            ABPack a, bb[4];
            ldpack(a, qrow, kk, koff);
#pragma unroll
            for (int j = 0; j < 4; ++j) ldpack(bb[j], krow[j], kk, koff);
#pragma unroll
            for (int j = 0; j < 4; ++j) c[j] = wmma16(a, bb[j], c[j]);
        }
#pragma unroll
        for (int r = 0; r < 8; ++r) {
            const int row = (r + hi * 8) * NS;
#pragma unroll
            for (int j = 0; j < 4; ++j)
                sc[row + (kt0 + j) * 16 + (lane & 15)] = c[j][r];
        }
    }
    __syncthreads();

    // ---- Phase 2: softmax rows in place (result stays f32 in sc) ----
    for (int r = wave; r < 16; r += 8) {
        float mx = -3.402823466e38f;
        for (int k = lane; k < NS; k += 32) mx = fmaxf(mx, sc[r * NS + k]);
#pragma unroll
        for (int off = 16; off > 0; off >>= 1) mx = fmaxf(mx, __shfl_xor(mx, off, 32));
        float sum = 0.f;
        for (int k = lane; k < NS; k += 32) {
            const float e = __expf(sc[r * NS + k] - mx);
            sc[r * NS + k] = e;
            sum += e;
        }
#pragma unroll
        for (int off = 16; off > 0; off >>= 1) sum += __shfl_xor(sum, off, 32);
        const float inv = 1.f / sum;
        for (int k = lane; k < NS; k += 32) sc[r * NS + k] *= inv;
    }
    __syncthreads();

    // ---- Phase 3: O[16][1024] = P[16][2048] @ V[2048][1024].
    // Each wave: 8 out-col tiles, 4 at a time; A built from f32 P with on-the-fly
    // cvt (co-executes with the matrix pipe).
    const float* prow = sc + (size_t)m * NS;
    for (int g = 0; g < 2; ++g) {
        const int nt0 = wave * 8 + g * 4;
        const _Float16* vrow[4];
#pragma unroll
        for (int j = 0; j < 4; ++j)
            vrow[j] = vt + ((size_t)b * NH + (nt0 + j) * 16 + m) * NS;
        v8f c[4] = {};
        for (int kk = 0; kk < NS; kk += 32) {
            ABPack a, bb[4];
            a.h[0] = cvt8(prow + kk + koff);
            a.h[1] = cvt8(prow + kk + 16 + koff);
#pragma unroll
            for (int j = 0; j < 4; ++j) ldpack(bb[j], vrow[j], kk, koff);
#pragma unroll
            for (int j = 0; j < 4; ++j) c[j] = wmma16(a, bb[j], c[j]);
        }
#pragma unroll
        for (int r = 0; r < 8; ++r) {
            const size_t row = ((size_t)b * NS + q0 + r + hi * 8) * NH;
#pragma unroll
            for (int j = 0; j < 4; ++j)
                out[row + (nt0 + j) * 16 + (lane & 15)] = c[j][r];
        }
    }
}

// ---------------------------------------------------------------------------
extern "C" void kernel_launch(void* const* d_in, const int* in_sizes, int n_in,
                              void* d_out, int out_size, void* d_ws, size_t ws_size,
                              hipStream_t stream) {
    (void)in_sizes; (void)n_in; (void)out_size; (void)ws_size;
    const float* x    = (const float*)d_in[0];   // lstm_out [B,S,H] f32
    const float* w    = (const float*)d_in[1];   // W [H,H] f32
    const float* bias = (const float*)d_in[2];   // b [H] f32
    float* out        = (float*)d_out;           // [B,S,H] f32

    const size_t nx = (size_t)NB * NS * NH;              // 16M elements
    _Float16* qk = (_Float16*)d_ws;                      // 32 MB
    _Float16* vt = qk + nx;                              // 32 MB
    _Float16* xh = vt + nx;                              // 32 MB
    _Float16* wh = xh + nx;                              //  2 MB

    // Kernel 0: one-time f32->f16 conversion of X and W
    cvt_kernel<<<(int)(nx / 4 / 256), 256, 0, stream>>>(x, xh, (int)(nx / 4));
    cvt_kernel<<<(NH * NH / 4 + 255) / 256, 256, 0, stream>>>(w, wh, NH * NH / 4);

    // Kernel 2: V transpose -> f16
    vt_kernel<<<dim3(NS / 32, NH / 32, NB), dim3(32, 8), 0, stream>>>(x, vt);

    // Kernel 1: linear projection -> f16 Q/K  (32x64 per wave, pure f16 operands)
    const int tiles = (NB * NS / 32) * (NH / 64);        // 8192 wave-tiles
    linear_qk_kernel<<<tiles / 8, 256, 0, stream>>>(xh, wh, bias, qk);

    // Kernel 3: attention (160 KB LDS -> 2 workgroups per 320KB WGP)
    const size_t smem = (size_t)16 * NH * 2    // Q tile f16
                      + (size_t)16 * NS * 4;   // scores / P f32
    hipFuncSetAttribute((const void*)attn_kernel,
                        hipFuncAttributeMaxDynamicSharedMemorySize, (int)smem);
    attn_kernel<<<NB * (NS / 16), 256, smem, stream>>>(qk, vt, out);
}
